// InvNet_office_24489903522664
// MI455X (gfx1250) — matrix operand order repaired and checked by reading, not compiled, not run on hardware
//
#include <hip/hip_runtime.h>
#include <math.h>

typedef __attribute__((ext_vector_type(16))) _Float16 v16h;
typedef __attribute__((ext_vector_type(8)))  _Float16 v8h;
typedef __attribute__((ext_vector_type(4)))  _Float16 v4h;
typedef __attribute__((ext_vector_type(8)))  float    v8f;
typedef __attribute__((ext_vector_type(4)))  float    v4f;

#define BATCH   128
#define NFEAT   2048
#define NCLS    16522
#define CTILE   128
#define NTILES  ((NCLS + CTILE - 1) / CTILE)   // 130
#define CPAD    (NTILES * CTILE)               // 16640
#define KCHUNK  64
#define LSTRIDE 72        // f16 units; 36 dwords/row -> conflict-free column reads
#define INV_BETA 20.0f
#define KNN     6

// ---------------------------------------------------------------------------
// Kernel 1: fused  (a) em -> new_em copy  (b) logits = inputs @ em.T / beta
// grid = 130 class tiles, block = 256 (8 waves). Wave w owns output rows
// 16w..16w+15 x all 128 classes of the tile. em chunk is converted to f16
// ONCE at staging time and double-buffered in LDS.
// ---------------------------------------------------------------------------
__global__ __launch_bounds__(256)
void gemm_logits_copy(const float* __restrict__ inputs,
                      const float* __restrict__ em,
                      float* __restrict__ new_em,    // d_out + 1 (4B-aligned only)
                      float* __restrict__ logits)    // ws, [128][CPAD]
{
    __shared__ _Float16 lds[2][CTILE * LSTRIDE];     // 2 x 18 KB
    const int tid  = threadIdx.x;
    const int wave = tid >> 5;
    const int lane = tid & 31;
    const int half = lane >> 4;
    const int l16  = lane & 15;
    const int cbase = blockIdx.x * CTILE;

    v8f acc[8] = {};
    v4f ld[8];                                       // in-flight chunk (reg staging)

    const int mrow = wave * 16 + l16;
    const float* arow = inputs + (size_t)mrow * NFEAT;

    // Cooperative chunk staging: thread covers 8 float4 slots of 128x64 f32.
    auto load_chunk = [&](int kb) {
        #pragma unroll
        for (int i = 0; i < 8; ++i) {
            int idx = i * 256 + tid;
            int r = idx >> 4, f4 = idx & 15;
            int c = cbase + r;
            ld[i] = {};
            if (c < NCLS)
                ld[i] = *(const v4f*)(em + (size_t)c * NFEAT + kb + f4 * 4);
        }
    };
    auto commit_chunk = [&](int buf, int kb) {       // new_em copy + f16 LDS store
        #pragma unroll
        for (int i = 0; i < 8; ++i) {
            int idx = i * 256 + tid;
            int r = idx >> 4, f4 = idx & 15;
            int c = cbase + r;
            if (c < NCLS) {
                float* dst = new_em + (size_t)c * NFEAT + kb + f4 * 4;
                dst[0] = ld[i].x; dst[1] = ld[i].y; dst[2] = ld[i].z; dst[3] = ld[i].w;
            }
            v4h h;
            h[0] = (_Float16)ld[i].x; h[1] = (_Float16)ld[i].y;
            h[2] = (_Float16)ld[i].z; h[3] = (_Float16)ld[i].w;
            *(v4h*)(&lds[buf][r * LSTRIDE + f4 * 4]) = h;
        }
    };

    load_chunk(0);
    commit_chunk(0, 0);
    __syncthreads();

    int buf = 0;
    for (int kb = 0; kb < NFEAT; kb += KCHUNK, buf ^= 1) {
        const int nkb = kb + KCHUNK;
        const bool has_next = nkb < NFEAT;
        if (has_next) load_chunk(nkb);               // HBM loads in flight over WMMA

        // Prefetch 2 chunks ahead: 256 threads x one 128B line = 32 KB chunk.
        if (nkb + KCHUNK < NFEAT) {
            int pr = cbase + (tid >> 1);
            if (pr < NCLS)
                __builtin_prefetch(em + (size_t)pr * NFEAT + nkb + KCHUNK + (tid & 1) * 32, 0, 1);
        }

        #pragma unroll
        for (int kk = 0; kk < KCHUNK; kk += 32) {
            // A fragment (16x32 f16): lanes 0-15 K 0..15, lanes 16-31 K 16..31.
            v16h a;
            {
                const float* p0 = arow + kb + kk + half * 16;
                const float* p1 = p0 + 8;
                #pragma unroll
                for (int j = 0; j < 8; ++j) {
                    a[j]     = (_Float16)p0[j];
                    a[j + 8] = (_Float16)p1[j];
                }
            }
            #pragma unroll
            for (int nt = 0; nt < 8; ++nt) {
                // B fragment (32x16 f16): N = lane%16, 16 contiguous K per lane.
                const _Float16* q = &lds[buf][(nt * 16 + l16) * LSTRIDE + kk + half * 16];
                v8h b0 = *(const v8h*)(q);           // 16B-aligned LDS loads
                v8h b1 = *(const v8h*)(q + 8);
                v16h b;
                #pragma unroll
                for (int j = 0; j < 8; ++j) { b[j] = b0[j]; b[j + 8] = b1[j]; }
                acc[nt] = __builtin_amdgcn_wmma_f32_16x16x32_f16(
                    false, a, false, b, (short)0, acc[nt], false, false);
            }
        }

        if (has_next) commit_chunk(buf ^ 1, nkb);    // fill the other buffer
        __syncthreads();
    }

    // C/D layout: VGPR j -> M = j (lanes 0-15) / j+8 (lanes 16-31), N = lane%16.
    #pragma unroll
    for (int nt = 0; nt < 8; ++nt) {
        int col = cbase + nt * 16 + l16;
        #pragma unroll
        for (int j = 0; j < 8; ++j) {
            int row = wave * 16 + (half ? (j + 8) : j);
            logits[(size_t)row * CPAD + col] = acc[nt][j] * INV_BETA;
        }
    }
}

// ---------------------------------------------------------------------------
// Kernel 2: per-row online log-sum-exp + top-6 + masked loss.
// grid = 128 rows, block = 256.
// ---------------------------------------------------------------------------
__global__ __launch_bounds__(256)
void row_reduce(const float* __restrict__ logits,
                const int* __restrict__ label,
                float* __restrict__ row_loss)
{
    __shared__ float sm[256], ss[256], sl[256];
    __shared__ float st[256 * KNN];
    const int tid = threadIdx.x;
    const int row = blockIdx.x;
    const float* lg = logits + (size_t)row * CPAD;
    const int lab = label[row];

    float m = -__builtin_inff(), s = 0.0f, lv = -__builtin_inff();
    float t[KNN];
    #pragma unroll
    for (int i = 0; i < KNN; ++i) t[i] = -__builtin_inff();

    for (int c = tid; c < NCLS; c += 256) {
        float x = lg[c];
        if (x > m) { s = s * __expf(m - x); m = x; }
        s += __expf(x - m);
        if (x > t[KNN - 1]) {
            t[KNN - 1] = x;
            #pragma unroll
            for (int i = KNN - 1; i > 0; --i)
                if (t[i] > t[i - 1]) { float u = t[i - 1]; t[i - 1] = t[i]; t[i] = u; }
        }
        if (c == lab) lv = x;
    }
    sm[tid] = m; ss[tid] = s; sl[tid] = lv;
    #pragma unroll
    for (int i = 0; i < KNN; ++i) st[tid * KNN + i] = t[i];
    __syncthreads();

    for (int off = 128; off > 0; off >>= 1) {
        if (tid < off) {
            float m1 = sm[tid], s1 = ss[tid];
            float m2 = sm[tid + off], s2 = ss[tid + off];
            float mm = fmaxf(m1, m2);
            ss[tid] = s1 * __expf(m1 - mm) + s2 * __expf(m2 - mm);
            sm[tid] = mm;
            sl[tid] = fmaxf(sl[tid], sl[tid + off]);
            float out[KNN];
            int i1 = 0, i2 = 0;
            #pragma unroll
            for (int i = 0; i < KNN; ++i) {
                float a = st[tid * KNN + i1];
                float b = st[(tid + off) * KNN + i2];
                if (a >= b) { out[i] = a; ++i1; } else { out[i] = b; ++i2; }
            }
            #pragma unroll
            for (int i = 0; i < KNN; ++i) st[tid * KNN + i] = out[i];
        }
        __syncthreads();
    }

    if (tid == 0) {
        float lse  = sm[0] + __logf(ss[0]);
        float sum6 = 0.0f;
        #pragma unroll
        for (int i = 0; i < KNN; ++i) sum6 += st[i];
        float kth = st[KNN - 1];
        float lb  = sl[0];
        bool inTop = (lb >= kth);
        float weighted = sum6 + lb + (inTop ? 0.0f : lb);  // label weight 2.0
        float mt = inTop ? 7.0f : 8.0f;
        row_loss[row] = -(weighted - mt * lse);
    }
}

// ---------------------------------------------------------------------------
// Kernel 3: finalize loss + sequential renormalized EMA on new_em.
// ---------------------------------------------------------------------------
__global__ __launch_bounds__(256)
void ema_finalize(const float* __restrict__ inputs,
                  const int* __restrict__ label,
                  const int* __restrict__ epoch,
                  const float* __restrict__ row_loss,
                  float* __restrict__ out)       // out[0]=loss, out+1 = new_em
{
    __shared__ float red[256];
    const int tid = threadIdx.x;

    red[tid] = (tid < BATCH) ? row_loss[tid] : 0.0f;
    __syncthreads();
    for (int off = 128; off > 0; off >>= 1) {
        if (tid < off) red[tid] += red[tid + off];
        __syncthreads();
    }
    if (tid == 0) out[0] = red[0] / (float)BATCH;

    float* nem = out + 1;
    const float alpha = 0.01f * (float)(*epoch);
    float vv[8];
    for (int i = 0; i < BATCH; ++i) {
        __syncthreads();
        const int y = label[i];
        float* emr = nem + (size_t)y * NFEAT + tid * 8;
        const float* xr = inputs + (size_t)i * NFEAT + tid * 8;
        float loc = 0.0f;
        #pragma unroll
        for (int j = 0; j < 8; ++j) {
            float v = alpha * emr[j] + (1.0f - alpha) * xr[j];
            vv[j] = v; loc += v * v;
        }
        red[tid] = loc;
        __syncthreads();
        for (int off = 128; off > 0; off >>= 1) {
            if (tid < off) red[tid] += red[tid + off];
            __syncthreads();
        }
        float rn = 1.0f / sqrtf(red[0]);
        #pragma unroll
        for (int j = 0; j < 8; ++j) emr[j] = vv[j] * rn;
    }
}

extern "C" void kernel_launch(void* const* d_in, const int* in_sizes, int n_in,
                              void* d_out, int out_size, void* d_ws, size_t ws_size,
                              hipStream_t stream) {
    const float* inputs = (const float*)d_in[0];
    const int*   label  = (const int*)d_in[1];
    const int*   epoch  = (const int*)d_in[2];
    const float* em     = (const float*)d_in[3];
    float* out = (float*)d_out;

    float* logits   = (float*)d_ws;                         // 128 * 16640 f32
    float* row_loss = logits + (size_t)BATCH * CPAD;        // 128 f32

    gemm_logits_copy<<<NTILES, 256, 0, stream>>>(inputs, em, out + 1, logits);
    row_reduce<<<BATCH, 256, 0, stream>>>(logits, label, row_loss);
    ema_finalize<<<1, 256, 0, stream>>>(inputs, label, epoch, row_loss, out);
}
